// RSSM_55259049230838
// MI455X (gfx1250) — matrix operand order repaired and checked by reading, not compiled, not run on hardware
//
#include <hip/hip_runtime.h>
#include <hip/hip_bf16.h>
#include <math.h>

typedef __bf16 bf16;
typedef bf16  bf16x8  __attribute__((ext_vector_type(8)));
typedef bf16  bf16x16 __attribute__((ext_vector_type(16)));
typedef float floatx8 __attribute__((ext_vector_type(8)));

#define WMMA_BF16(a, b, c) \
  __builtin_amdgcn_wmma_f32_16x16x32_bf16(false, (a), false, (b), (short)0, (c), false, false)

static constexpr int Ldim = 1024, Sdim = 256, Adim = 32, Rdim = 512, Tdim = 64, Bdim = 512;

// ---------------- fragment loaders (bf16 activations, bf16 weights) ----------------
// A fragment (16x32 bf16) per ISA layout: lanes 0-15: M=lane, K={k0..k0+7} U {k0+16..k0+23};
// lanes 16-31: M=lane-16, K={k0+8..k0+15} U {k0+24..k0+31}. Two aligned 16B loads per lane.
__device__ __forceinline__ bf16x16 cat16(bf16x8 lo, bf16x8 hi) {
  return __builtin_shufflevector(lo, hi, 0, 1, 2, 3, 4, 5, 6, 7, 8, 9, 10, 11, 12, 13, 14, 15);
}

__device__ __forceinline__ bf16x16 ldA_bf(const bf16* X, int ld, int m0, int k0, int lane) {
  const bf16* p = X + (size_t)(m0 + (lane & 15)) * ld + k0 + ((lane >> 4) << 3);
  return cat16(*(const bf16x8*)p, *(const bf16x8*)(p + 16));
}

// 8-element chunk from a virtual concat [X0 | X1]; split is a multiple of 8 so no straddle.
__device__ __forceinline__ bf16x8 ld8cat(const bf16* X0, int ld0, const bf16* X1, int ld1,
                                         int split, int m, int k) {
  const bf16* p = (k < split) ? X0 + (size_t)m * ld0 + k
                              : X1 + (size_t)m * ld1 + (k - split);
  return *(const bf16x8*)p;
}

__device__ __forceinline__ bf16x16 ldA_cat_bf(const bf16* X0, int ld0, const bf16* X1, int ld1,
                                              int split, int m0, int k0, int lane) {
  int m  = m0 + (lane & 15);
  int kb = k0 + ((lane >> 4) << 3);
  return cat16(ld8cat(X0, ld0, X1, ld1, split, m, kb),
               ld8cat(X0, ld0, X1, ld1, split, m, kb + 16));
}

// B fragment (32x16 bf16) from row-major N x K weight: lane<16 holds col N=n0+lane,
// K={k0..k0+7} U {k0+16..k0+23}; lanes 16-31 hold K={k0+8..k0+15} U {k0+24..k0+31}.
__device__ __forceinline__ bf16x16 ldB(const bf16* W, int ld, int n0, int k0, int lane) {
  int n  = n0 + (lane & 15);
  int kb = k0 + ((lane >> 4) << 3);
  const bf16* p = W + (size_t)n * ld + kb;
  return cat16(*(const bf16x8*)p, *(const bf16x8*)(p + 16));
}

__device__ __forceinline__ float sigmoidf_(float x) { return 1.0f / (1.0f + __expf(-x)); }
__device__ __forceinline__ float softplusf_(float x) {
  return (x > 20.0f) ? x : log1pf(__expf(x));
}

// ---------------- fp32 -> bf16 convert ----------------
__global__ void k_cvt(const float* __restrict__ src, bf16* __restrict__ dst, int n) {
  int i = blockIdx.x * blockDim.x + threadIdx.x;
  if (i < n) dst[i] = (bf16)src[i];
}

// ---------------- kernel 1: hidden = relu(concat(state, a_t) @ Wtp^T + btp) -> bf16 ----------
__global__ __launch_bounds__(256)
void k_trans_pre(const bf16* __restrict__ state_bf, const bf16* __restrict__ act_bf,
                 const bf16* __restrict__ Wtp, const float* __restrict__ btp,
                 bf16* __restrict__ hidden_bf) {
  int lane  = threadIdx.x & 31;
  int gwave = blockIdx.x * (blockDim.x >> 5) + (threadIdx.x >> 5);
  if (gwave >= 32 * 64) return;
  int m0 = (gwave >> 6) * 16, n0 = (gwave & 63) * 16;
  floatx8 acc = {};
  for (int k0 = 0; k0 < Sdim + Adim; k0 += 32) {
    bf16x16 a = ldA_cat_bf(state_bf, Sdim, act_bf, Adim, Sdim, m0, k0, lane);
    bf16x16 b = ldB(Wtp, Sdim + Adim, n0, k0, lane);
    acc = WMMA_BF16(a, b, acc);
  }
  int n = n0 + (lane & 15);
  float bias = btp[n];
#pragma unroll
  for (int v = 0; v < 8; ++v) {
    int m = m0 + v + ((lane >> 4) << 3);
    float h = acc[v] + bias;
    hidden_bf[(size_t)m * Ldim + n] = (bf16)(h > 0.0f ? h : 0.0f);
  }
}

// ---------------- kernel 2: fused GRU cell, 2 M-tiles per wave ----------------
__global__ __launch_bounds__(256)
void k_gru(const bf16* __restrict__ hidden_bf, const bf16* __restrict__ latp_bf,
           const float* __restrict__ latp_f32,
           const bf16* __restrict__ Wih, const float* __restrict__ bih,
           const bf16* __restrict__ Whh, const float* __restrict__ bhh,
           float* __restrict__ lat_out, bf16* __restrict__ lat_out_bf) {
  int lane  = threadIdx.x & 31;
  int gwave = blockIdx.x * (blockDim.x >> 5) + (threadIdx.x >> 5);
  if (gwave >= 16 * 64) return;
  int n0 = (gwave & 63) * 16;
  int m0 = (gwave >> 6) * 32;  // two M tiles: m0, m0+16
  floatx8 acc_i[2][3] = {{{}, {}, {}}, {{}, {}, {}}};
  floatx8 acc_h[2][3] = {{{}, {}, {}}, {{}, {}, {}}};
  for (int k0 = 0; k0 < Ldim; k0 += 32) {
    bf16x16 ah0 = ldA_bf(hidden_bf, Ldim, m0, k0, lane);
    bf16x16 ah1 = ldA_bf(hidden_bf, Ldim, m0 + 16, k0, lane);
    bf16x16 al0 = ldA_bf(latp_bf, Ldim, m0, k0, lane);
    bf16x16 al1 = ldA_bf(latp_bf, Ldim, m0 + 16, k0, lane);
    if (k0 + 32 < Ldim) {  // prefetch next K chunk of this lane's weight rows
      __builtin_prefetch(Wih + (size_t)(n0 + (lane & 15)) * Ldim + k0 + 32, 0, 1);
      __builtin_prefetch(Whh + (size_t)(n0 + (lane & 15)) * Ldim + k0 + 32, 0, 1);
    }
#pragma unroll
    for (int g = 0; g < 3; ++g) {
      bf16x16 bi = ldB(Wih, Ldim, n0 + g * Ldim, k0, lane);
      acc_i[0][g] = WMMA_BF16(ah0, bi, acc_i[0][g]);
      acc_i[1][g] = WMMA_BF16(ah1, bi, acc_i[1][g]);
      bf16x16 bh = ldB(Whh, Ldim, n0 + g * Ldim, k0, lane);
      acc_h[0][g] = WMMA_BF16(al0, bh, acc_h[0][g]);
      acc_h[1][g] = WMMA_BF16(al1, bh, acc_h[1][g]);
    }
  }
  int n = n0 + (lane & 15);
  float bir = bih[n], biz = bih[Ldim + n], bin = bih[2 * Ldim + n];
  float bhr = bhh[n], bhz = bhh[Ldim + n], bhn = bhh[2 * Ldim + n];
#pragma unroll
  for (int mi = 0; mi < 2; ++mi) {
#pragma unroll
    for (int v = 0; v < 8; ++v) {
      int m = m0 + mi * 16 + v + ((lane >> 4) << 3);
      float r  = sigmoidf_(acc_i[mi][0][v] + bir + acc_h[mi][0][v] + bhr);
      float z  = sigmoidf_(acc_i[mi][1][v] + biz + acc_h[mi][1][v] + bhz);
      float nn = tanhf(acc_i[mi][2][v] + bin + r * (acc_h[mi][2][v] + bhn));
      float hprev = latp_f32[(size_t)m * Ldim + n];
      float hnew  = (1.0f - z) * nn + z * hprev;
      lat_out[(size_t)m * Ldim + n]    = hnew;
      lat_out_bf[(size_t)m * Ldim + n] = (bf16)hnew;
    }
  }
}

// ---------------- kernel 3: posterior + rsample ----------------
__global__ __launch_bounds__(256)
void k_post(const bf16* __restrict__ hidden_bf, const bf16* __restrict__ Wpo,
            const float* __restrict__ bpo, const float* __restrict__ eps_t,
            float* __restrict__ mean_o, float* __restrict__ std_o,
            float* __restrict__ samp_o, bf16* __restrict__ samp_bf) {
  int lane  = threadIdx.x & 31;
  int gwave = blockIdx.x * (blockDim.x >> 5) + (threadIdx.x >> 5);
  if (gwave >= 32 * 16) return;
  int m0 = (gwave >> 4) * 16, n0 = (gwave & 15) * 16;
  floatx8 acc_m = {}, acc_s = {};
  for (int k0 = 0; k0 < Ldim; k0 += 32) {
    bf16x16 a  = ldA_bf(hidden_bf, Ldim, m0, k0, lane);
    bf16x16 bm = ldB(Wpo, Ldim, n0, k0, lane);
    acc_m = WMMA_BF16(a, bm, acc_m);
    bf16x16 bs = ldB(Wpo, Ldim, Sdim + n0, k0, lane);
    acc_s = WMMA_BF16(a, bs, acc_s);
  }
  int n = n0 + (lane & 15);
  float bm = bpo[n], bs = bpo[Sdim + n];
#pragma unroll
  for (int v = 0; v < 8; ++v) {
    int m = m0 + v + ((lane >> 4) << 3);
    size_t idx = (size_t)m * Sdim + n;
    float mean = acc_m[v] + bm;
    float sd   = softplusf_(acc_s[v] + bs);
    float sp   = mean + sd * eps_t[idx];
    mean_o[idx] = mean;
    std_o[idx]  = sd;
    samp_o[idx] = sp;
    samp_bf[idx] = (bf16)sp;
  }
}

// ---------------- kernel 4: x1 = relu(concat(latent, sampled) @ Wr1^T + br1) -> bf16 ----------
__global__ __launch_bounds__(256)
void k_r1(const bf16* __restrict__ lat_bf, const bf16* __restrict__ samp_bf,
          const bf16* __restrict__ Wr1, const float* __restrict__ br1,
          bf16* __restrict__ x1_bf) {
  int lane  = threadIdx.x & 31;
  int gwave = blockIdx.x * (blockDim.x >> 5) + (threadIdx.x >> 5);
  if (gwave >= 32 * 32) return;
  int m0 = (gwave >> 5) * 16, n0 = (gwave & 31) * 16;
  floatx8 acc = {};
  for (int k0 = 0; k0 < Ldim + Sdim; k0 += 32) {
    bf16x16 a = ldA_cat_bf(lat_bf, Ldim, samp_bf, Sdim, Ldim, m0, k0, lane);
    bf16x16 b = ldB(Wr1, Ldim + Sdim, n0, k0, lane);
    acc = WMMA_BF16(a, b, acc);
  }
  int n = n0 + (lane & 15);
  float bias = br1[n];
#pragma unroll
  for (int v = 0; v < 8; ++v) {
    int m = m0 + v + ((lane >> 4) << 3);
    float h = acc[v] + bias;
    x1_bf[(size_t)m * Rdim + n] = (bf16)(h > 0.0f ? h : 0.0f);
  }
}

// ---------------- kernel 5: x2 = relu(x1 @ Wr2^T + br2) -> fp32 (for GEMV) ----------------
__global__ __launch_bounds__(256)
void k_r2(const bf16* __restrict__ x1_bf, const bf16* __restrict__ Wr2,
          const float* __restrict__ br2, float* __restrict__ x2) {
  int lane  = threadIdx.x & 31;
  int gwave = blockIdx.x * (blockDim.x >> 5) + (threadIdx.x >> 5);
  if (gwave >= 32 * 32) return;
  int m0 = (gwave >> 5) * 16, n0 = (gwave & 31) * 16;
  floatx8 acc = {};
  for (int k0 = 0; k0 < Rdim; k0 += 32) {
    bf16x16 a = ldA_bf(x1_bf, Rdim, m0, k0, lane);
    bf16x16 b = ldB(Wr2, Rdim, n0, k0, lane);
    acc = WMMA_BF16(a, b, acc);
  }
  int n = n0 + (lane & 15);
  float bias = br2[n];
#pragma unroll
  for (int v = 0; v < 8; ++v) {
    int m = m0 + v + ((lane >> 4) << 3);
    float h = acc[v] + bias;
    x2[(size_t)m * Rdim + n] = h > 0.0f ? h : 0.0f;
  }
}

// ---------------- kernel 6: reward GEMV (N=1), wave per row, fp32 ----------------
__global__ __launch_bounds__(256)
void k_r3(const float* __restrict__ x2, const float* __restrict__ Wr3,
          const float* __restrict__ br3, float* __restrict__ rew) {
  int lane = threadIdx.x & 31;
  int row  = blockIdx.x * (blockDim.x >> 5) + (threadIdx.x >> 5);
  if (row >= Bdim) return;
  const float* xr = x2 + (size_t)row * Rdim;
  float s = 0.0f;
  for (int k = lane; k < Rdim; k += 32) s += xr[k] * Wr3[k];
#pragma unroll
  for (int off = 16; off > 0; off >>= 1) s += __shfl_down(s, off, 32);
  if (lane == 0) rew[row] = s + br3[0];
}

// ---------------- host launch ----------------
extern "C" void kernel_launch(void* const* d_in, const int* in_sizes, int n_in,
                              void* d_out, int out_size, void* d_ws, size_t ws_size,
                              hipStream_t stream) {
  const float* prev_state = (const float*)d_in[0];
  const float* actions    = (const float*)d_in[1];
  const float* prev_lat   = (const float*)d_in[2];
  const float* eps        = (const float*)d_in[3];
  const float* W_ih = (const float*)d_in[4];   const float* b_ih = (const float*)d_in[5];
  const float* W_hh = (const float*)d_in[6];   const float* b_hh = (const float*)d_in[7];
  const float* W_tp = (const float*)d_in[8];   const float* b_tp = (const float*)d_in[9];
  const float* W_po = (const float*)d_in[10];  const float* b_po = (const float*)d_in[11];
  const float* W_r1 = (const float*)d_in[12];  const float* b_r1 = (const float*)d_in[13];
  const float* W_r2 = (const float*)d_in[14];  const float* b_r2 = (const float*)d_in[15];
  const float* W_r3 = (const float*)d_in[16];  const float* b_r3 = (const float*)d_in[17];

  float* out = (float*)d_out;
  float* latents = out;                                        // T*B*L
  float* states  = latents + (size_t)Tdim * Bdim * Ldim;       // T*B*S
  float* means   = states  + (size_t)Tdim * Bdim * Sdim;
  float* stds    = means   + (size_t)Tdim * Bdim * Sdim;
  float* rewards = stds    + (size_t)Tdim * Bdim * Sdim;       // T*B

  // workspace bump allocator (256B aligned)
  char* ws = (char*)d_ws;
  size_t off = 0;
  auto alloc = [&](size_t bytes) {
    void* p = ws + off;
    off += (bytes + 255) & ~size_t(255);
    return p;
  };
  bf16* Wih_b  = (bf16*)alloc((size_t)3 * Ldim * Ldim * sizeof(bf16));
  bf16* Whh_b  = (bf16*)alloc((size_t)3 * Ldim * Ldim * sizeof(bf16));
  bf16* Wtp_b  = (bf16*)alloc((size_t)Ldim * (Sdim + Adim) * sizeof(bf16));
  bf16* Wpo_b  = (bf16*)alloc((size_t)2 * Sdim * Ldim * sizeof(bf16));
  bf16* Wr1_b  = (bf16*)alloc((size_t)Rdim * (Ldim + Sdim) * sizeof(bf16));
  bf16* Wr2_b  = (bf16*)alloc((size_t)Rdim * Rdim * sizeof(bf16));
  bf16* act_bf = (bf16*)alloc((size_t)Tdim * Bdim * Adim * sizeof(bf16));
  bf16* st0_bf = (bf16*)alloc((size_t)Bdim * Sdim * sizeof(bf16));
  bf16* lat_bf0 = (bf16*)alloc((size_t)Bdim * Ldim * sizeof(bf16));   // ping
  bf16* lat_bf1 = (bf16*)alloc((size_t)Bdim * Ldim * sizeof(bf16));   // pong
  bf16* hid_bf  = (bf16*)alloc((size_t)Bdim * Ldim * sizeof(bf16));
  bf16* samp_bf = (bf16*)alloc((size_t)Bdim * Sdim * sizeof(bf16));
  bf16* x1_bf   = (bf16*)alloc((size_t)Bdim * Rdim * sizeof(bf16));
  float* x2     = (float*)alloc((size_t)Bdim * Rdim * sizeof(float));
  (void)ws_size;

  auto cvt = [&](const float* s, bf16* d, int n) {
    k_cvt<<<(n + 255) / 256, 256, 0, stream>>>(s, d, n);
  };
  cvt(W_ih, Wih_b, 3 * Ldim * Ldim);
  cvt(W_hh, Whh_b, 3 * Ldim * Ldim);
  cvt(W_tp, Wtp_b, Ldim * (Sdim + Adim));
  cvt(W_po, Wpo_b, 2 * Sdim * Ldim);
  cvt(W_r1, Wr1_b, Rdim * (Ldim + Sdim));
  cvt(W_r2, Wr2_b, Rdim * Rdim);
  cvt(actions, act_bf, Tdim * Bdim * Adim);
  cvt(prev_state, st0_bf, Bdim * Sdim);
  cvt(prev_lat, lat_bf0, Bdim * Ldim);

  bf16* lat_bf[2] = {lat_bf0, lat_bf1};

  for (int t = 0; t < Tdim; ++t) {
    const bf16*  state_bf  = (t == 0) ? st0_bf : samp_bf;
    const bf16*  latp_bf   = lat_bf[t & 1];
    bf16*        latn_bf   = lat_bf[(t + 1) & 1];
    const float* latp_f32  = (t == 0) ? prev_lat : latents + (size_t)(t - 1) * Bdim * Ldim;
    const bf16*  act_t     = act_bf + (size_t)t * Bdim * Adim;
    const float* eps_t     = eps + (size_t)t * Bdim * Sdim;
    float* lat_t  = latents + (size_t)t * Bdim * Ldim;
    float* samp_t = states  + (size_t)t * Bdim * Sdim;
    float* mean_t = means   + (size_t)t * Bdim * Sdim;
    float* std_t  = stds    + (size_t)t * Bdim * Sdim;
    float* rew_t  = rewards + (size_t)t * Bdim;

    k_trans_pre<<<256, 256, 0, stream>>>(state_bf, act_t, Wtp_b, b_tp, hid_bf);
    k_gru<<<128, 256, 0, stream>>>(hid_bf, latp_bf, latp_f32, Wih_b, b_ih, Whh_b, b_hh,
                                   lat_t, latn_bf);
    k_post<<<64, 256, 0, stream>>>(hid_bf, Wpo_b, b_po, eps_t, mean_t, std_t, samp_t, samp_bf);
    k_r1<<<128, 256, 0, stream>>>(latn_bf, samp_bf, Wr1_b, b_r1, x1_bf);
    k_r2<<<128, 256, 0, stream>>>(x1_bf, Wr2_b, b_r2, x2);
    k_r3<<<64, 256, 0, stream>>>(x2, W_r3, b_r3, rew_t);
  }
}
